// HamiHead_uuw_84748294684798
// MI455X (gfx1250) — matrix operand order repaired and checked by reading, not compiled, not run on hardware
//
#include <hip/hip_runtime.h>
#include <math.h>

typedef __attribute__((ext_vector_type(16))) _Float16 v16h;
typedef __attribute__((ext_vector_type(8)))  float    v8f;
typedef __attribute__((ext_vector_type(2)))  _Float16 h2v;

#define N_NODE 2560
#define N_EDGE 48640
#define HS 128
#define NPW 1792
#define NINS 19

// ---- instruction tables (li, row-block j, col-block k, mul1,l1, mul2,l2, offsets) ----
__constant__ int c_li[NINS]   = {0,0,0, 1,1,1,1,1,1, 2,2,2,2,2,2, 3,3,3, 4};
__constant__ int c_jb[NINS]   = {0,1,2, 0,1,1,1,2,2, 0,1,1,2,2,2, 1,2,2, 2};
__constant__ int c_kb[NINS]   = {0,1,2, 1,0,1,2,1,2, 2,1,2,0,1,2, 2,1,2, 2};
__constant__ int c_m1[NINS]   = {3,2,1, 3,2,2,2,1,1, 3,2,2,1,1,1, 2,1,1, 1};
__constant__ int c_l1[NINS]   = {0,1,2, 0,1,1,1,2,2, 0,1,1,2,2,2, 1,2,2, 2};
__constant__ int c_m2[NINS]   = {3,2,1, 2,3,2,1,2,1, 1,2,1,3,2,1, 1,2,1, 1};
__constant__ int c_l2[NINS]   = {0,1,2, 1,0,1,2,1,2, 2,1,2,0,1,2, 2,1,2, 2};
__constant__ int c_woff[NINS] = {0,288,416, 448,640,832,960,1024,1088,
                                 1120,1216,1344,1408,1504,1568, 1600,1664,1728, 1760};
__constant__ int c_boff[NINS] = {0,9,13, 0,0,0,0,0,0, 0,0,0,0,0,0, 0,0,0, 0};
__constant__ int c_w3off[NINS]= {0,1,10, 35,44,53,80,125,170,
                                 245,270,315,390,415,490, 615,720,825, 1000};
__constant__ int c_inoff[5]   = {0,32,128,288,512};
__constant__ int c_rowoff[3]  = {0,3,9};

// ------------------- f32 -> f16 convert -------------------
__global__ void f2h_kernel(const float* __restrict__ src, _Float16* __restrict__ dst, int n) {
    int i = blockIdx.x * blockDim.x + threadIdx.x;
    if (i < n) dst[i] = (_Float16)src[i];
}

// ------------------- Wigner-3j (e3nn convention) on device, fp64 -------------------
__device__ double dfact(int n) { double r = 1.0; for (int i = 2; i <= n; ++i) r *= (double)i; return r; }

__device__ void su2cg(int j1, int j2, int j3, double* C) {
    int d2 = 2*j2+1, d3 = 2*j3+1, d1 = 2*j1+1;
    for (int i = 0; i < d1*d2*d3; ++i) C[i] = 0.0;
    for (int m1 = -j1; m1 <= j1; ++m1)
    for (int m2 = -j2; m2 <= j2; ++m2) {
        int m3 = m1 + m2;
        if (m3 < -j3 || m3 > j3) continue;
        int vmin = 0;
        if (-j1 + j2 + m3 > vmin) vmin = -j1 + j2 + m3;
        if (-j1 + m1 > vmin) vmin = -j1 + m1;
        int vmax = j2 + j3 + m1;
        if (j3 - j1 + j2 < vmax) vmax = j3 - j1 + j2;
        if (j3 + m3 < vmax) vmax = j3 + m3;
        if (vmax < vmin) continue;
        double norm = sqrt((2.0*j3 + 1.0) *
            dfact(j3+j1-j2)*dfact(j3-j1+j2)*dfact(j1+j2-j3)*dfact(j3+m3)*dfact(j3-m3) /
            (dfact(j1+j2+j3+1)*dfact(j1-m1)*dfact(j1+m1)*dfact(j2-m2)*dfact(j2+m2)));
        double S = 0.0;
        for (int v = vmin; v <= vmax; ++v) {
            double sg = ((v + j2 + m2) & 1) ? -1.0 : 1.0;
            S += sg * dfact(j2+j3+m1-v)*dfact(j1-m1+v) /
                 (dfact(v)*dfact(j3-j1+j2-v)*dfact(j3+m3-v)*dfact(v+j1-j2-m3));
        }
        C[((j1+m1)*d2 + (j2+m2))*d3 + (j3+m3)] = norm * S;
    }
}

__device__ void buildQ(int l, double* qr, double* qi) {
    int d = 2*l+1;
    for (int i = 0; i < d*d; ++i) { qr[i] = 0.0; qi[i] = 0.0; }
    const double s = 0.70710678118654752440;
    for (int m = -l; m < 0; ++m) {
        qr[(l+m)*d + (l-m)] = s;       // 1/sqrt2
        qi[(l+m)*d + (l+m)] = -s;      // -i/sqrt2
    }
    qr[l*d + l] = 1.0;
    for (int m = 1; m <= l; ++m) {
        double sg = (m & 1) ? -1.0 : 1.0;
        qr[(l+m)*d + (l+m)] = sg * s;
        qi[(l+m)*d + (l-m)] = sg * s;  // i*(-1)^m/sqrt2
    }
    int ph = l & 3;  // (-i)^l
    double pr = (ph==0) ? 1.0 : (ph==2) ? -1.0 : 0.0;
    double pi = (ph==1) ? -1.0 : (ph==3) ? 1.0 : 0.0;
    for (int i = 0; i < d*d; ++i) {
        double r = qr[i], im = qi[i];
        qr[i] = r*pr - im*pi;
        qi[i] = r*pi + im*pr;
    }
}

__global__ void w3j_kernel(float* __restrict__ out) {
    if (threadIdx.x != 0 || blockIdx.x != 0) return;
    double C[225], R[225];
    double q1r[25], q1i[25], q2r[25], q2i[25], q3r[81], q3i[81];
    for (int t = 0; t < NINS; ++t) {
        int l1 = c_l1[t], l2 = c_l2[t], l3 = c_li[t];
        int d1 = 2*l1+1, d2 = 2*l2+1, d3 = 2*l3+1;
        su2cg(l1, l2, l3, C);
        buildQ(l1, q1r, q1i);
        buildQ(l2, q2r, q2i);
        buildQ(l3, q3r, q3i);
        double nrm = 0.0;
        for (int j = 0; j < d1; ++j)
        for (int l = 0; l < d2; ++l)
        for (int m = 0; m < d3; ++m) {
            double sr = 0.0;
            for (int i = 0; i < d1; ++i)
            for (int k = 0; k < d2; ++k)
            for (int n = 0; n < d3; ++n) {
                double cv = C[(i*d2 + k)*d3 + n];
                if (cv == 0.0) continue;
                double ar = q1r[i*d1+j]*q2r[k*d2+l] - q1i[i*d1+j]*q2i[k*d2+l];
                double ai = q1r[i*d1+j]*q2i[k*d2+l] + q1i[i*d1+j]*q2r[k*d2+l];
                // multiply by conj(Q3[n][m])
                double br = ar*q3r[n*d3+m] + ai*q3i[n*d3+m];
                sr += cv * br;
            }
            R[(j*d2 + l)*d3 + m] = sr;
            nrm += sr * sr;
        }
        nrm = sqrt(nrm);
        for (int i = 0; i < d1*d2*d3; ++i)
            out[c_w3off[t] + i] = (float)(R[i] / nrm);
    }
}

// ------------------- hidden = silu(A @ W1^T + b1) , f16 WMMA, K in {128,256} -------------------
__global__ __launch_bounds__(256)
void hidden_kernel(const _Float16* __restrict__ node16,
                   const int* __restrict__ dstIdx, const int* __restrict__ srcIdx,
                   int pairMode, int K,
                   const _Float16* __restrict__ W1, const float* __restrict__ b1,
                   _Float16* __restrict__ out) {
    int wave = threadIdx.x >> 5, lane = threadIdx.x & 31;
    int hh = lane >> 4, nn = lane & 15;
    int row0 = blockIdx.x * 16;
    int n0 = wave * 16;
    int g = row0 + nn;  // A row handled by this lane (M = lane%16)
    const _Float16* arow0;
    const _Float16* arow1 = nullptr;
    if (pairMode) {
        arow0 = node16 + (size_t)dstIdx[g] * HS;
        arow1 = node16 + (size_t)srcIdx[g] * HS;
    } else {
        arow0 = node16 + (size_t)g * HS;
    }
    const _Float16* brow = W1 + (size_t)(n0 + nn) * K;
    v8f c = {0.f, 0.f, 0.f, 0.f, 0.f, 0.f, 0.f, 0.f};
    for (int kk = 0; kk < K; kk += 32) {
        v16h a, b;
#pragma unroll
        for (int ep = 0; ep < 8; ++ep) {
            int ka = kk + 2*ep + 8*hh + ((ep >= 4) ? 8 : 0);
            h2v pa;
            if (pairMode) {
                const _Float16* base = (ka < HS) ? arow0 : (arow1 - HS);
                pa = *(const h2v*)(base + ka);
            } else {
                pa = *(const h2v*)(arow0 + ka);
            }
            a[2*ep] = pa[0]; a[2*ep+1] = pa[1];
            int kb = kk + 2*ep + 8*hh + ((ep >= 4) ? 8 : 0);
            h2v pb = *(const h2v*)(brow + kb);
            b[2*ep] = pb[0]; b[2*ep+1] = pb[1];
        }
        c = __builtin_amdgcn_wmma_f32_16x16x32_f16(false, a, false, b, (short)0, c, false, false);
    }
    float bias = b1[n0 + nn];
#pragma unroll
    for (int r = 0; r < 8; ++r) {
        float v = c[r] + bias;
        // silu via fast hardware reciprocal (v_rcp_f32), avoids IEEE div expansion
        float s = v * __builtin_amdgcn_rcpf(1.f + __expf(-v));
        out[(size_t)(row0 + r + 8*hh) * HS + n0 + nn] = (_Float16)s;
    }
}

// ------------------- fused weights-GEMM + expansion -------------------
__global__ __launch_bounds__(256)
void expansion_kernel(const float* __restrict__ x,           // [B,800]
                      const _Float16* __restrict__ hW,       // [B,128]
                      const _Float16* __restrict__ hB,       // [B,128]
                      const _Float16* __restrict__ W2,       // [1792,128]
                      const float* __restrict__ b2,          // [1792]
                      const _Float16* __restrict__ W2b,      // [14,128]
                      const float* __restrict__ b2b,         // [14]
                      const float* __restrict__ w3j,         // [1225]
                      float* __restrict__ Hout) {            // [B,196]
    __shared__ float xs[16 * 288];
    __shared__ float wch[16 * 288];
    __shared__ float res_s[16 * 20];
    __shared__ float Hs[16 * 196];
    __shared__ float bias_s[16 * 14];

    const int tid = threadIdx.x;
    const int row0 = blockIdx.x * 16;

    for (int i = tid; i < 16 * 196; i += 256) Hs[i] = 0.f;
    for (int i = tid; i < 16 * 14; i += 256) {
        int r = i / 14, o = i % 14;
        float acc = b2b[o];
        const _Float16* hb = hB + (size_t)(row0 + r) * HS;
        const _Float16* wr = W2b + (size_t)o * HS;
        for (int cc = 0; cc < HS; ++cc) acc += (float)hb[cc] * (float)wr[cc];
        bias_s[i] = acc;
    }
    __syncthreads();

    const int wave = tid >> 5, lane = tid & 31;
    const int hh = lane >> 4, nn = lane & 15;
    const _Float16* arow = hW + (size_t)(row0 + nn) * HS;

    // Preload the block-invariant A tile (hidden 16x128) as 4 WMMA fragments (32 VGPRs).
    v16h afrag[4];
#pragma unroll
    for (int kq = 0; kq < 4; ++kq) {
#pragma unroll
        for (int ep = 0; ep < 8; ++ep) {
            int ka = 32*kq + 2*ep + 8*hh + ((ep >= 4) ? 8 : 0);
            h2v pa = *(const h2v*)(arow + ka);
            afrag[kq][2*ep] = pa[0]; afrag[kq][2*ep+1] = pa[1];
        }
    }

    int t = 0;
    for (int gli = 0; gli < 5; ++gli) {
        int din = 2 * gli + 1;
        int cnt = 16 * 32 * din;
        int ioff = c_inoff[gli];
        for (int i = tid; i < cnt; i += 256) {
            int r = i / (32 * din), rem = i % (32 * din);
            xs[r * (32 * din) + rem] = x[(size_t)(row0 + r) * 800 + ioff + rem];
        }
        __syncthreads();
        while (t < NINS && c_li[t] == gli) {
            const int mm1 = c_m1[t], mm2 = c_m2[t];
            const int m1m2 = mm1 * mm2;
            const int nw = 32 * m1m2, nst = nw / 16;
            const int wo = c_woff[t];
            // ---- weights chunk: [16 rows] x [nw] = hidden(16x128) @ W2_chunk^T via WMMA ----
            for (int st = wave; st < nst; st += 8) {
                const _Float16* brow = W2 + (size_t)(wo + st * 16 + nn) * HS;
                v8f c = {0.f, 0.f, 0.f, 0.f, 0.f, 0.f, 0.f, 0.f};
#pragma unroll
                for (int kq = 0; kq < 4; ++kq) {
                    v16h b;
#pragma unroll
                    for (int ep = 0; ep < 8; ++ep) {
                        int kb = 32*kq + 2*ep + 8*hh + ((ep >= 4) ? 8 : 0);
                        h2v pb = *(const h2v*)(brow + kb);
                        b[2*ep] = pb[0]; b[2*ep+1] = pb[1];
                    }
                    c = __builtin_amdgcn_wmma_f32_16x16x32_f16(false, afrag[kq], false, b,
                                                               (short)0, c, false, false);
                }
                float bb = b2[wo + st * 16 + nn];
#pragma unroll
                for (int r = 0; r < 8; ++r)
                    wch[(r + 8*hh) * nw + st * 16 + nn] = c[r] + bb;
            }
            __syncthreads();
            // ---- res[u][v][k] = sum_w w[w,u,v] * x1[w,k] (+ bias if li==0) ----
            const int szr = m1m2 * din;
            for (int i = tid; i < 16 * szr; i += 256) {
                int r = i / szr, rem = i % szr;
                int uv = rem / din, kk2 = rem % din;
                float acc = 0.f;
                for (int w = 0; w < 32; ++w)
                    acc += wch[r * nw + w * m1m2 + uv] * xs[r * (32 * din) + w * din + kk2];
                if (gli == 0) acc += bias_s[r * 14 + c_boff[t] + uv];
                res_s[i] = acc;
            }
            __syncthreads();
            // ---- blk[u,i,v,j] = sum_k W3J[i,j,k]*res[u,v,k] / 32 -> accumulate into H ----
            const int d1 = 2 * c_l1[t] + 1, d2 = 2 * c_l2[t] + 1;
            const int per = mm1 * d1 * mm2 * d2;
            const int ro = c_rowoff[c_jb[t]], co = c_rowoff[c_kb[t]];
            const float* wj = w3j + c_w3off[t];
            for (int i = tid; i < 16 * per; i += 256) {
                int r = i / per, rem = i % per;
                int u = rem / (d1 * mm2 * d2); rem %= (d1 * mm2 * d2);
                int i2 = rem / (mm2 * d2); rem %= (mm2 * d2);
                int v = rem / d2;
                int j2 = rem % d2;
                float s = 0.f;
                for (int kk2 = 0; kk2 < din; ++kk2)
                    s += wj[(i2 * d2 + j2) * din + kk2] * res_s[r * szr + (u * mm2 + v) * din + kk2];
                Hs[r * 196 + (ro + u * d1 + i2) * 14 + (co + v * d2 + j2)] += s * (1.f / 32.f);
            }
            __syncthreads();
            ++t;
        }
    }
    for (int i = tid; i < 16 * 196; i += 256)
        Hout[(size_t)row0 * 196 + i] = Hs[i];
}

// ------------------- in-place symmetrization -------------------
__global__ void symm_diag_kernel(float* __restrict__ H) {
    int total = N_NODE * 105;
    for (int idx = blockIdx.x * blockDim.x + threadIdx.x; idx < total;
         idx += gridDim.x * blockDim.x) {
        int b = idx / 105, p = idx % 105;
        int r = 0;
        while (p >= 14 - r) { p -= 14 - r; ++r; }
        int c = r + p;
        float* M = H + (size_t)b * 196;
        float a = M[r * 14 + c], bb = M[c * 14 + r];
        float v = 0.5f * (a + bb);
        M[r * 14 + c] = v;
        M[c * 14 + r] = v;
    }
}

__global__ void symm_nd_kernel(float* __restrict__ H, const int* __restrict__ tei) {
    int total = N_EDGE * 196;
    for (int idx = blockIdx.x * blockDim.x + threadIdx.x; idx < total;
         idx += gridDim.x * blockDim.x) {
        int e = idx / 196, q = idx % 196;
        int te = tei[e];
        if (te < e) continue;  // partner thread handles the pair
        int r = q / 14, c = q % 14;
        float a = H[(size_t)e * 196 + r * 14 + c];
        float bt = H[(size_t)te * 196 + c * 14 + r];
        float v = 0.5f * (a + bt);
        H[(size_t)e * 196 + r * 14 + c] = v;
        H[(size_t)te * 196 + c * 14 + r] = v;
    }
}

// ------------------- host launcher -------------------
extern "C" void kernel_launch(void* const* d_in, const int* in_sizes, int n_in,
                              void* d_out, int out_size, void* d_ws, size_t ws_size,
                              hipStream_t stream) {
    const float* node_attr = (const float*)d_in[0];
    const float* fii       = (const float*)d_in[1];
    const float* fij       = (const float*)d_in[2];
    const float* w1ii  = (const float*)d_in[3];
    const float* b1ii  = (const float*)d_in[4];
    const float* w2ii  = (const float*)d_in[5];
    const float* b2ii  = (const float*)d_in[6];
    const float* w1iib = (const float*)d_in[7];
    const float* b1iib = (const float*)d_in[8];
    const float* w2iib = (const float*)d_in[9];
    const float* b2iib = (const float*)d_in[10];
    const float* w1ij  = (const float*)d_in[11];
    const float* b1ij  = (const float*)d_in[12];
    const float* w2ij  = (const float*)d_in[13];
    const float* b2ij  = (const float*)d_in[14];
    const float* w1ijb = (const float*)d_in[15];
    const float* b1ijb = (const float*)d_in[16];
    const float* w2ijb = (const float*)d_in[17];
    const float* b2ijb = (const float*)d_in[18];
    const int* fei = (const int*)d_in[19];   // [2,E]: dst row then src row
    const int* tei = (const int*)d_in[20];
    float* out = (float*)d_out;

    // workspace carve-out (all f16 chunks have even element counts -> alignment OK)
    _Float16* p = (_Float16*)d_ws;
    _Float16* node16  = p; p += (size_t)N_NODE * HS;
    _Float16* w1ii16  = p; p += 128 * 128;
    _Float16* w1iib16 = p; p += 128 * 128;
    _Float16* w2ii16  = p; p += NPW * 128;
    _Float16* w2iib16 = p; p += 14 * 128;
    _Float16* w1ij16  = p; p += 128 * 256;
    _Float16* w1ijb16 = p; p += 128 * 256;
    _Float16* w2ij16  = p; p += NPW * 128;
    _Float16* w2ijb16 = p; p += 14 * 128;
    _Float16* h_ii    = p; p += (size_t)N_NODE * HS;
    _Float16* h_iib   = p; p += (size_t)N_NODE * HS;
    _Float16* h_ij    = p; p += (size_t)N_EDGE * HS;
    _Float16* h_ijb   = p; p += (size_t)N_EDGE * HS;
    float* w3jbuf = (float*)p;

    // f32 -> f16 conversions
    struct Cv { const float* s; _Float16* d; int n; };
    const Cv cv[] = {
        {node_attr, node16, N_NODE * HS},
        {w1ii, w1ii16, 128 * 128}, {w1iib, w1iib16, 128 * 128},
        {w2ii, w2ii16, NPW * 128}, {w2iib, w2iib16, 14 * 128},
        {w1ij, w1ij16, 128 * 256}, {w1ijb, w1ijb16, 128 * 256},
        {w2ij, w2ij16, NPW * 128}, {w2ijb, w2ijb16, 14 * 128},
    };
    for (int i = 0; i < 9; ++i)
        f2h_kernel<<<(cv[i].n + 255) / 256, 256, 0, stream>>>(cv[i].s, cv[i].d, cv[i].n);

    w3j_kernel<<<1, 64, 0, stream>>>(w3jbuf);

    const int* dstIdx = fei;
    const int* srcIdx = fei + N_EDGE;

    hidden_kernel<<<N_NODE / 16, 256, 0, stream>>>(node16, nullptr, nullptr, 0, 128,
                                                   w1ii16, b1ii, h_ii);
    hidden_kernel<<<N_NODE / 16, 256, 0, stream>>>(node16, nullptr, nullptr, 0, 128,
                                                   w1iib16, b1iib, h_iib);
    hidden_kernel<<<N_EDGE / 16, 256, 0, stream>>>(node16, dstIdx, srcIdx, 1, 256,
                                                   w1ij16, b1ij, h_ij);
    hidden_kernel<<<N_EDGE / 16, 256, 0, stream>>>(node16, dstIdx, srcIdx, 1, 256,
                                                   w1ijb16, b1ijb, h_ijb);

    expansion_kernel<<<N_NODE / 16, 256, 0, stream>>>(fii, h_ii, h_iib, w2ii16, b2ii,
                                                      w2iib16, b2iib, w3jbuf, out);
    expansion_kernel<<<N_EDGE / 16, 256, 0, stream>>>(fij, h_ij, h_ijb, w2ij16, b2ij,
                                                      w2ijb16, b2ijb, w3jbuf,
                                                      out + (size_t)N_NODE * 196);

    symm_diag_kernel<<<(N_NODE * 105 + 255) / 256, 256, 0, stream>>>(out);
    symm_nd_kernel<<<(N_EDGE * 196 + 255) / 256, 256, 0, stream>>>(out + (size_t)N_NODE * 196, tei);
}
